// FewShotNCALoss_39273180954624
// MI455X (gfx1250) — matrix operand order repaired and checked by reading, not compiled, not run on hardware
//
#include <hip/hip_runtime.h>
#include <hip/hip_bf16.h>

typedef __attribute__((ext_vector_type(16))) __bf16 v16bf;
typedef __attribute__((ext_vector_type(8)))  float  v8f;

#define NPTS 8192
#define DIM  256
#define EPSV 1e-10f

struct U8x4 { uint4 a, b; };   // 32 bytes == 16 bf16

static __device__ __forceinline__ unsigned short f32_to_bf16_rne(float x) {
    unsigned int u = __float_as_uint(x);
    unsigned int r = u + 0x7FFFu + ((u >> 16) & 1u);
    return (unsigned short)(r >> 16);
}
static __device__ __forceinline__ float bf16u_to_f32(unsigned short h) {
    return __uint_as_float(((unsigned int)h) << 16);
}

// ---------------------------------------------------------------------------
// Phase 1: fp32 -> (bf16 hi, bf16 lo) split + per-row squared norms.
// ---------------------------------------------------------------------------
__global__ void nca_prep_kernel(const float* __restrict__ pred,
                                unsigned short* __restrict__ phi,
                                unsigned short* __restrict__ plo,
                                float* __restrict__ sq) {
    const int row = blockIdx.x;
    const int t   = threadIdx.x;
    const float x = pred[(size_t)row * DIM + t];
    const unsigned short h = f32_to_bf16_rne(x);
    const float xf = bf16u_to_f32(h);
    const unsigned short l = f32_to_bf16_rne(x - xf);
    phi[(size_t)row * DIM + t] = h;
    plo[(size_t)row * DIM + t] = l;

    __shared__ float red[256];
    red[t] = x * x;
    __syncthreads();
    for (int s = 128; s > 0; s >>= 1) {
        if (t < s) red[t] += red[t + s];
        __syncthreads();
    }
    if (t == 0) sq[row] = red[0];
}

// ---------------------------------------------------------------------------
// Phase 2: fused Gram (bf16 split-WMMA) -> exp -> masked column sums.
// Block = 16 output columns; 8 waves; wave w owns row tiles t*128+w*16.
// A panels stream global -> LDS via async copies (ASYNCcnt, 4-step pipeline,
// wave-private LDS buffers, no barriers), then LDS -> VGPR -> WMMA.
// B panel (K=256, hi+lo) register-resident; 3 independent WMMA chains.
// ---------------------------------------------------------------------------
__global__ void __launch_bounds__(256, 1)
nca_gram_kernel(const unsigned short* __restrict__ phi,
                const unsigned short* __restrict__ plo,
                const float* __restrict__ sq,
                const int* __restrict__ target,
                const float* __restrict__ tempPtr,
                float* __restrict__ gnum,
                float* __restrict__ gden) {
    // 8 waves x 4 buffers x 2KB (hi0|hi1|lo0|lo1, 512B per chunk, lane*16)
    __shared__ uint4 smem4[4096];                     // 64 KB staging
    __shared__ float sNum[16], sDen[16];

    const int lane   = threadIdx.x & 31;
    const int wave   = threadIdx.x >> 5;
    const int laneN  = lane & 15;
    const int hiHalf = lane >> 4;          // 0: lanes 0-15, 1: lanes 16-31
    const int cb     = blockIdx.x * 16;    // column tile base

    const float invT = 1.0f / tempPtr[0];

    // Per-lane column constants (column j fixed for entire kernel)
    const int   j   = cb + laneN;
    const float sqj = sq[j];
    const int   tj  = target[j];

    // Preload B panel: two 16B runs (k=8h.., 16+8h..) of pred[cb+laneN] per K-step.
    v16bf Bhi[8], Blo[8];
    {
        const unsigned short* bh = phi + (size_t)j * DIM + hiHalf * 8;
        const unsigned short* bl = plo + (size_t)j * DIM + hiHalf * 8;
#pragma unroll
        for (int s = 0; s < 8; ++s) {
            U8x4 u, v;
            u.a = *(const uint4*)(bh + s * 32);
            u.b = *(const uint4*)(bh + s * 32 + 16);
            v.a = *(const uint4*)(bl + s * 32);
            v.b = *(const uint4*)(bl + s * 32 + 16);
            Bhi[s] = __builtin_bit_cast(v16bf, u);
            Blo[s] = __builtin_bit_cast(v16bf, v);
        }
    }

    // LDS addressing for async copies (VDST is a byte offset into LDS)
    const unsigned ldsBase = (unsigned)(size_t)(&smem4[0]);
    const unsigned sbyte   = ldsBase + (unsigned)wave * 8192u + (unsigned)lane * 16u;
    const int      sidx    = wave * 512 + lane;      // uint4 index into smem4

    // Per-lane global byte pointers for the A stream.
    // Step q (q = t*8 + s): row = t*128 + wave*16 + laneN,
    //   byte = row*512 + hiHalf*16 + s*64  (two 16B chunks at +0 / +32)
    unsigned long long ihg = (unsigned long long)(size_t)phi +
        (unsigned long long)(wave * 16 + laneN) * 512ull + (unsigned)hiHalf * 16u;
    unsigned long long ilg = (unsigned long long)(size_t)plo +
        (unsigned long long)(wave * 16 + laneN) * 512ull + (unsigned)hiHalf * 16u;

    // inst offset applies to BOTH global and LDS addresses, so chunk1's vdst
    // is pre-biased by -32 to land at its 512B-aligned LDS slot.
    auto issueStep = [&](int buf) {
        const unsigned vd0 = sbyte + (unsigned)buf * 2048u;
        asm volatile("global_load_async_to_lds_b128 %0, %1, off"
                     :: "v"(vd0), "v"(ihg) : "memory");
        asm volatile("global_load_async_to_lds_b128 %0, %1, off offset:32"
                     :: "v"(vd0 + 480u), "v"(ihg) : "memory");
        asm volatile("global_load_async_to_lds_b128 %0, %1, off"
                     :: "v"(vd0 + 1024u), "v"(ilg) : "memory");
        asm volatile("global_load_async_to_lds_b128 %0, %1, off offset:32"
                     :: "v"(vd0 + 1504u), "v"(ilg) : "memory");
    };

    // Prologue: prime the pipeline with steps 0,1,2 (buffers 0,1,2).
    issueStep(0); ihg += 64; ilg += 64;
    issueStep(1); ihg += 64; ilg += 64;
    issueStep(2); ihg += 64; ilg += 64;

    float numAcc = 0.0f, denAcc = 0.0f;

    for (int t = 0; t < 64; ++t) {
        const int  rowbase = t * 128 + wave * 16;
        const bool lastT   = (t == 63);

        v8f chh = {0.f, 0.f, 0.f, 0.f, 0.f, 0.f, 0.f, 0.f};
        v8f chl = {0.f, 0.f, 0.f, 0.f, 0.f, 0.f, 0.f, 0.f};
        v8f clh = {0.f, 0.f, 0.f, 0.f, 0.f, 0.f, 0.f, 0.f};

#pragma unroll
        for (int s = 0; s < 8; ++s) {
            // Step q = t*8+s in-flight window is q..q+2 (12 async ops);
            // in-order completion => waiting to <=8 guarantees step q landed.
            if (!lastT || s < 6)  asm volatile("s_wait_asynccnt 8" ::: "memory");
            else if (s == 6)      asm volatile("s_wait_asynccnt 4" ::: "memory");
            else                  asm volatile("s_wait_asynccnt 0" ::: "memory");

            const int buf = s & 3;               // (t*8+s) % 4 == s % 4
            U8x4 uh, ul;
            uh.a = smem4[sidx + buf * 128];
            uh.b = smem4[sidx + buf * 128 + 32];
            ul.a = smem4[sidx + buf * 128 + 64];
            ul.b = smem4[sidx + buf * 128 + 96];
            v16bf Ahi = __builtin_bit_cast(v16bf, uh);
            v16bf Alo = __builtin_bit_cast(v16bf, ul);

            // split-precision dot: hi*hi + hi*lo + lo*hi, 3 independent chains
            chh = __builtin_amdgcn_wmma_f32_16x16x32_bf16(false, Ahi, false, Bhi[s], (short)0, chh, false, false);
            chl = __builtin_amdgcn_wmma_f32_16x16x32_bf16(false, Ahi, false, Blo[s], (short)0, chl, false, false);
            clh = __builtin_amdgcn_wmma_f32_16x16x32_bf16(false, Alo, false, Bhi[s], (short)0, clh, false, false);

            // Issue step q+3 into buffer (s+3)&3 (its previous occupant was
            // consumed -- and DScnt-waited -- in the previous iteration).
            if (t < 63 || s < 5) {
                issueStep((s + 3) & 3);
                const unsigned long long adv = (((s + 3) & 7) == 7) ? 65088ull : 64ull;
                ihg += adv; ilg += adv;
            }
        }

        // Epilogue: C VGPR r holds (M = r + 8*hiHalf, N = laneN).
        const float4 s0 = *(const float4*)(sq + rowbase + hiHalf * 8);
        const float4 s1 = *(const float4*)(sq + rowbase + hiHalf * 8 + 4);
        const int4   t0 = *(const int4*)(target + rowbase + hiHalf * 8);
        const int4   t1 = *(const int4*)(target + rowbase + hiHalf * 8 + 4);
        const int ibase = rowbase + hiHalf * 8;

        auto elem = [&](float dot, float si, int ti, int i) {
            float p = si + sqj - 2.0f * dot;
            p = fmaxf(p, EPSV);
            const float d = __expf(-p * invT);
            const bool same = (ti == tj);
            numAcc += (same && (i != j)) ? d : 0.0f;   // positives = same - eye
            denAcc += (!same) ? d : 0.0f;              // negatives
        };
        elem(chh[0] + chl[0] + clh[0], s0.x, t0.x, ibase + 0);
        elem(chh[1] + chl[1] + clh[1], s0.y, t0.y, ibase + 1);
        elem(chh[2] + chl[2] + clh[2], s0.z, t0.z, ibase + 2);
        elem(chh[3] + chl[3] + clh[3], s0.w, t0.w, ibase + 3);
        elem(chh[4] + chl[4] + clh[4], s1.x, t1.x, ibase + 4);
        elem(chh[5] + chl[5] + clh[5], s1.y, t1.y, ibase + 5);
        elem(chh[6] + chl[6] + clh[6], s1.z, t1.z, ibase + 6);
        elem(chh[7] + chl[7] + clh[7], s1.w, t1.w, ibase + 7);
    }

    // Reduce the 16 columns across lane halves + 8 waves via LDS atomics.
    if (threadIdx.x < 16) { sNum[threadIdx.x] = 0.0f; sDen[threadIdx.x] = 0.0f; }
    __syncthreads();
    atomicAdd(&sNum[laneN], numAcc);
    atomicAdd(&sDen[laneN], denAcc);
    __syncthreads();
    if (threadIdx.x < 16) {
        gnum[cb + threadIdx.x] = sNum[threadIdx.x];
        gden[cb + threadIdx.x] = sDen[threadIdx.x];
    }
}

// ---------------------------------------------------------------------------
// Phase 3: frac = num/(num+max(den,eps)); loss = -mean(log(frac) | frac>=eps)
// ---------------------------------------------------------------------------
__global__ void nca_finalize_kernel(const float* __restrict__ gnum,
                                    const float* __restrict__ gden,
                                    float* __restrict__ out) {
    __shared__ float red[256];
    float acc = 0.0f;
    for (int jj = threadIdx.x; jj < NPTS; jj += 256) {
        const float nu = gnum[jj];
        const float de = fmaxf(gden[jj], EPSV);
        const float frac = nu / (nu + de);
        acc += (frac >= EPSV) ? -__logf(frac) : 0.0f;
    }
    red[threadIdx.x] = acc;
    __syncthreads();
    for (int s = 128; s > 0; s >>= 1) {
        if (threadIdx.x < s) red[threadIdx.x] += red[threadIdx.x + s];
        __syncthreads();
    }
    if (threadIdx.x == 0) out[0] = red[0] / (float)NPTS;
}

extern "C" void kernel_launch(void* const* d_in, const int* in_sizes, int n_in,
                              void* d_out, int out_size, void* d_ws, size_t ws_size,
                              hipStream_t stream) {
    (void)in_sizes; (void)n_in; (void)out_size; (void)ws_size;
    const float* pred   = (const float*)d_in[0];
    const int*   target = (const int*)d_in[1];
    const float* temp   = (const float*)d_in[2];

    char* ws = (char*)d_ws;
    unsigned short* phi = (unsigned short*)ws;                                 // 4 MB
    unsigned short* plo = (unsigned short*)(ws + (size_t)NPTS * DIM * 2);      // 4 MB
    float* sq   = (float*)(ws + (size_t)NPTS * DIM * 4);                       // 32 KB
    float* gnum = sq + NPTS;                                                   // 32 KB
    float* gden = gnum + NPTS;                                                 // 32 KB

    nca_prep_kernel<<<NPTS, DIM, 0, stream>>>(pred, phi, plo, sq);
    nca_gram_kernel<<<NPTS / 16, 256, 0, stream>>>(phi, plo, sq, target, temp, gnum, gden);
    nca_finalize_kernel<<<1, 256, 0, stream>>>(gnum, gden, (float*)d_out);
}
